// GNO_61366492725767
// MI455X (gfx1250) — matrix operand order repaired and checked by reading, not compiled, not run on hardware
//
#include <hip/hip_runtime.h>
#include <hip/hip_bf16.h>
#include <math.h>

typedef __attribute__((ext_vector_type(16))) _Float16 v16h;
typedef __attribute__((ext_vector_type(8)))  float    v8f;

#define N_NODES 100000
#define N_EDGES 1600000
#define DLAT 64

__device__ __forceinline__ float gelu_exact(float x) {
    return 0.5f * x * (1.0f + erff(x * 0.7071067811865475f));
}

// ---- WMMA fragment helpers (CDNA5 16x16x32 f16, wave32) ----

// A fragment (16xK f16, MxK) from LDS row-major [16][ldk], K-chunk base k0.
// ISA 16-bit A 16x32 layout: lanes 0-15: V0..3 K=0..7, V4..7 K=16..23;
// lanes 16-31: V0..3 K=8..15, V4..7 K=24..31. row M = lane&15.
__device__ __forceinline__ v16h load_a_frag(const _Float16* s, int ldk, int k0) {
    int lane = threadIdx.x & 31;
    int row  = lane & 15;
    int kb   = (lane < 16) ? 0 : 8;
    v16h a;
#pragma unroll
    for (int e = 0; e < 16; ++e) {
        int v = e >> 1, lo = e & 1;
        int k = (v < 4) ? (kb + 2 * v + lo) : (16 + kb + 2 * (v - 4) + lo);
        a[e] = s[row * ldk + k0 + k];
    }
    return a;
}

// B fragment (32x16 f16, KxN) from global row-major [K][64] fp16 weights.
// lanes 0-15: K=0..7,16..23, lanes 16-31: K=8..15,24..31 (same as A). col N = lane&15.
__device__ __forceinline__ v16h load_b_frag(const _Float16* w, int k0, int n0) {
    int lane = threadIdx.x & 31;
    int n  = n0 + (lane & 15);
    int kb = k0 + ((lane < 16) ? 0 : 8);
    v16h b;
#pragma unroll
    for (int e = 0; e < 16; ++e) b[e] = w[(kb + ((e < 8) ? e : (8 + e))) * DLAT + n];
    return b;
}

// D (f32 16x16) -> LDS as f16 at column base n0. M = v + 8*(lane>=16), N = lane&15.
__device__ __forceinline__ void store_d_f16(v8f c, _Float16* s, int ld, int n0) {
    int lane = threadIdx.x & 31;
    int n  = n0 + (lane & 15);
    int mb = (lane < 16) ? 0 : 8;
#pragma unroll
    for (int v = 0; v < 8; ++v) s[(mb + v) * ld + n] = (_Float16)c[v];
}

// ---- Kernel 1: weight conversion to fp16 (pad p_w1 K:18 -> 32) ----
__global__ void __launch_bounds__(256)
prep_weights_kernel(const float* __restrict__ p_w1, const float* __restrict__ p_w2,
                    const float* __restrict__ w_w1, const float* __restrict__ w_w2,
                    const float* __restrict__ d_w1,
                    _Float16* __restrict__ W1p, _Float16* __restrict__ W2p,
                    _Float16* __restrict__ Ww1p, _Float16* __restrict__ Ww2p,
                    _Float16* __restrict__ Dw1p) {
    int stride = gridDim.x * blockDim.x;
    int t = blockIdx.x * blockDim.x + threadIdx.x;
    for (int i = t; i < 32 * 64; i += stride) {
        int k = i >> 6;
        W1p[i] = (_Float16)((k < 18) ? p_w1[i] : 0.0f);
    }
    for (int i = t; i < 64 * 64; i += stride) {
        W2p[i]  = (_Float16)p_w2[i];
        Ww1p[i] = (_Float16)w_w1[i];
        Ww2p[i] = (_Float16)w_w2[i];
        Dw1p[i] = (_Float16)d_w1[i];
    }
}

// ---- Kernel 2: zero the aggregation buffer ----
__global__ void __launch_bounds__(256)
zero_aggr_kernel(float* __restrict__ aggr) {
    int i = blockIdx.x * blockDim.x + threadIdx.x;
    if (i * 4 < N_NODES * DLAT) {
        float4 z = {0.f, 0.f, 0.f, 0.f};
        ((float4*)aggr)[i] = z;
    }
}

// ---- Kernel 3: project MLP (18 -> 64 -> 64), fused per 16-node wave tile ----
__global__ void __launch_bounds__(256)
project_kernel(const float* __restrict__ x, const float* __restrict__ grid,
               const float* __restrict__ p_b1, const float* __restrict__ p_b2,
               const _Float16* __restrict__ W1p, const _Float16* __restrict__ W2p,
               _Float16* __restrict__ Hh) {
    __shared__ _Float16 stage[8][16 * 64];
    int wave = threadIdx.x >> 5;
    int lane = threadIdx.x & 31;
    _Float16* s = stage[wave];
    int base = (blockIdx.x * 8 + wave) * 16;

    // Stage input tile: K 0..15 = x, K 16,17 = grid, K 18..31 = 0.
    {
        int f0 = lane * 8;                       // 8 x-floats per lane, same row
        int row = base + (f0 >> 4);
        bool ok = row < N_NODES;
#pragma unroll
        for (int j = 0; j < 8; ++j) {
            int ff = f0 + j;
            float vv = ok ? x[base * 16 + ff] : 0.0f;
            s[(ff >> 4) * 64 + (ff & 15)] = (_Float16)vv;
        }
        int row2 = base + (lane >> 1);
        float g = (row2 < N_NODES) ? grid[base * 2 + lane] : 0.0f;
        s[(lane >> 1) * 64 + 16 + (lane & 1)] = (_Float16)g;
#pragma unroll
        for (int j = 0; j < 7; ++j)
            s[(lane >> 1) * 64 + 18 + (lane & 1) + 2 * j] = (_Float16)0.0f;
    }
    __syncthreads();

    // Layer 1: [16x32] @ [32x64] + b1, exact GELU.
    v16h a0 = load_a_frag(s, 64, 0);
    v8f acc[4];
#pragma unroll
    for (int t = 0; t < 4; ++t) {
        v16h b = load_b_frag(W1p, 0, t * 16);
        v8f c = {};
        c = __builtin_amdgcn_wmma_f32_16x16x32_f16(false, a0, false, b, (short)0, c, false, false);
        float bias = p_b1[t * 16 + (lane & 15)];
#pragma unroll
        for (int v = 0; v < 8; ++v) acc[t][v] = gelu_exact(c[v] + bias);
    }
    __syncthreads();
#pragma unroll
    for (int t = 0; t < 4; ++t) store_d_f16(acc[t], s, 64, t * 16);
    __syncthreads();

    // Layer 2: [16x64] @ [64x64] + b2 (linear output) -> Hh fp16.
#pragma unroll
    for (int t = 0; t < 4; ++t) {
        v8f c = {};
#pragma unroll
        for (int kc = 0; kc < 2; ++kc) {
            v16h a = load_a_frag(s, 64, kc * 32);
            v16h b = load_b_frag(W2p, kc * 32, t * 16);
            c = __builtin_amdgcn_wmma_f32_16x16x32_f16(false, a, false, b, (short)0, c, false, false);
        }
        float bias = p_b2[t * 16 + (lane & 15)];
        int n  = t * 16 + (lane & 15);
        int mb = (lane < 16) ? 0 : 8;
#pragma unroll
        for (int v = 0; v < 8; ++v) {
            int row = base + mb + v;
            if (row < N_NODES) Hh[(size_t)row * DLAT + n] = (_Float16)(c[v] + bias);
        }
    }
}

// ---- Kernel 4: edge scatter-add: aggr[dst] += h[src] (L2-resident atomics) ----
__global__ void __launch_bounds__(256)
scatter_kernel(const int* __restrict__ ei, const _Float16* __restrict__ Hh,
               float* __restrict__ aggr) {
    long long tid = (long long)blockIdx.x * 256 + threadIdx.x;
    if (tid >= (long long)N_EDGES * 16) return;
    int e  = (int)(tid >> 4);
    int d4 = ((int)tid & 15) * 4;
    int src = ei[e];
    int dst = ei[N_EDGES + e];
    const _Float16* hp = Hh + (size_t)src * DLAT + d4;
    float* ap = aggr + (size_t)dst * DLAT + d4;
#pragma unroll
    for (int j = 0; j < 4; ++j) atomicAdd(ap + j, (float)hp[j]);
}

// ---- Kernel 5: update (W-MLP + aggr + GELU) and decode (64->64->1) ----
__global__ void __launch_bounds__(256)
update_decode_kernel(const _Float16* __restrict__ Hh, const float* __restrict__ aggr,
                     const float* __restrict__ w_b1, const float* __restrict__ w_b2,
                     const float* __restrict__ d_b1, const float* __restrict__ d_w2v,
                     const float* __restrict__ d_b2,
                     const _Float16* __restrict__ Ww1p, const _Float16* __restrict__ Ww2p,
                     const _Float16* __restrict__ Dw1p,
                     float* __restrict__ out) {
    __shared__ _Float16 stage[8][16 * 64];
    __shared__ float   fstage[8][16 * 64];
    int wave = threadIdx.x >> 5;
    int lane = threadIdx.x & 31;
    _Float16* s = stage[wave];
    float*    f = fstage[wave];
    int base = (blockIdx.x * 8 + wave) * 16;

    // Stage h tile (1024 contiguous halves) and aggr tile (1024 contiguous floats).
    {
        int row = base + (lane >> 1);
        bool ok = row < N_NODES;
        uint4* dh = (uint4*)s + lane * 4;
        if (ok) {
            const uint4* g = (const uint4*)(Hh + (size_t)base * DLAT) + lane * 4;
#pragma unroll
            for (int j = 0; j < 4; ++j) dh[j] = g[j];
        } else {
            uint4 z = {0, 0, 0, 0};
#pragma unroll
            for (int j = 0; j < 4; ++j) dh[j] = z;
        }
        float4* df = (float4*)f + lane * 8;
        if (ok) {
            const float4* g = (const float4*)(aggr + (size_t)base * DLAT) + lane * 8;
#pragma unroll
            for (int j = 0; j < 8; ++j) df[j] = g[j];
        } else {
            float4 z = {0.f, 0.f, 0.f, 0.f};
#pragma unroll
            for (int j = 0; j < 8; ++j) df[j] = z;
        }
    }
    __syncthreads();

    // W-MLP layer 1: gelu(h @ w_w1 + b1)
    v8f acc[4];
#pragma unroll
    for (int t = 0; t < 4; ++t) {
        v8f c = {};
#pragma unroll
        for (int kc = 0; kc < 2; ++kc) {
            v16h a = load_a_frag(s, 64, kc * 32);
            v16h b = load_b_frag(Ww1p, kc * 32, t * 16);
            c = __builtin_amdgcn_wmma_f32_16x16x32_f16(false, a, false, b, (short)0, c, false, false);
        }
        float bias = w_b1[t * 16 + (lane & 15)];
#pragma unroll
        for (int v = 0; v < 8; ++v) acc[t][v] = gelu_exact(c[v] + bias);
    }
    __syncthreads();
#pragma unroll
    for (int t = 0; t < 4; ++t) store_d_f16(acc[t], s, 64, t * 16);
    __syncthreads();

    // W-MLP layer 2 + aggr, then gelu -> u
#pragma unroll
    for (int t = 0; t < 4; ++t) {
        v8f c = {};
#pragma unroll
        for (int kc = 0; kc < 2; ++kc) {
            v16h a = load_a_frag(s, 64, kc * 32);
            v16h b = load_b_frag(Ww2p, kc * 32, t * 16);
            c = __builtin_amdgcn_wmma_f32_16x16x32_f16(false, a, false, b, (short)0, c, false, false);
        }
        float bias = w_b2[t * 16 + (lane & 15)];
        int n  = t * 16 + (lane & 15);
        int mb = (lane < 16) ? 0 : 8;
#pragma unroll
        for (int v = 0; v < 8; ++v)
            acc[t][v] = gelu_exact(c[v] + bias + f[(mb + v) * 64 + n]);
    }
    __syncthreads();
#pragma unroll
    for (int t = 0; t < 4; ++t) store_d_f16(acc[t], s, 64, t * 16);
    __syncthreads();

    // Decode layer 1: gelu(u @ d_w1 + b1) -> fstage (f32)
#pragma unroll
    for (int t = 0; t < 4; ++t) {
        v8f c = {};
#pragma unroll
        for (int kc = 0; kc < 2; ++kc) {
            v16h a = load_a_frag(s, 64, kc * 32);
            v16h b = load_b_frag(Dw1p, kc * 32, t * 16);
            c = __builtin_amdgcn_wmma_f32_16x16x32_f16(false, a, false, b, (short)0, c, false, false);
        }
        float bias = d_b1[t * 16 + (lane & 15)];
        int n  = t * 16 + (lane & 15);
        int mb = (lane < 16) ? 0 : 8;
#pragma unroll
        for (int v = 0; v < 8; ++v)
            f[(mb + v) * 64 + n] = gelu_exact(c[v] + bias);
    }
    __syncthreads();

    // Decode layer 2: per-node 64-wide dot + d_b2 (VALU).
    if (lane < 16) {
        int row = base + lane;
        if (row < N_NODES) {
            float sacc = d_b2[0];
#pragma unroll
            for (int k = 0; k < 64; ++k) sacc += f[lane * 64 + k] * d_w2v[k];
            out[row] = sacc;
        }
    }
}

extern "C" void kernel_launch(void* const* d_in, const int* in_sizes, int n_in,
                              void* d_out, int out_size, void* d_ws, size_t ws_size,
                              hipStream_t stream) {
    const float* x    = (const float*)d_in[0];
    const float* grid = (const float*)d_in[1];
    const int*   ei   = (const int*)d_in[2];
    // d_in[3] edge_features: unused by the reference message function
    const float* p_w1 = (const float*)d_in[4];
    const float* p_b1 = (const float*)d_in[5];
    const float* p_w2 = (const float*)d_in[6];
    const float* p_b2 = (const float*)d_in[7];
    const float* w_w1 = (const float*)d_in[8];
    const float* w_b1 = (const float*)d_in[9];
    const float* w_w2 = (const float*)d_in[10];
    const float* w_b2 = (const float*)d_in[11];
    const float* d_w1 = (const float*)d_in[12];
    const float* d_b1 = (const float*)d_in[13];
    const float* d_w2 = (const float*)d_in[14];
    const float* d_b2 = (const float*)d_in[15];
    float* out = (float*)d_out;

    // Workspace layout (256B aligned):
    char* ws = (char*)d_ws;
    _Float16* W1p  = (_Float16*)(ws + 0);         // 32*64 f16  = 4096 B
    _Float16* W2p  = (_Float16*)(ws + 4096);      // 64*64 f16  = 8192 B
    _Float16* Ww1p = (_Float16*)(ws + 12288);
    _Float16* Ww2p = (_Float16*)(ws + 20480);
    _Float16* Dw1p = (_Float16*)(ws + 28672);
    _Float16* Hh   = (_Float16*)(ws + 36864);     // N*64 f16 = 12.8 MB
    float*    AGGR = (float*)(ws + 36864 + (size_t)N_NODES * DLAT * 2); // N*64 f32 = 25.6 MB

    prep_weights_kernel<<<32, 256, 0, stream>>>(p_w1, p_w2, w_w1, w_w2, d_w1,
                                                W1p, W2p, Ww1p, Ww2p, Dw1p);
    zero_aggr_kernel<<<(N_NODES * DLAT / 4 + 255) / 256, 256, 0, stream>>>(AGGR);

    int nblk = (N_NODES + 127) / 128;  // 8 waves/block, 16 nodes/wave
    project_kernel<<<nblk, 256, 0, stream>>>(x, grid, p_b1, p_b2, W1p, W2p, Hh);

    long long sthreads = (long long)N_EDGES * 16;
    scatter_kernel<<<(int)((sthreads + 255) / 256), 256, 0, stream>>>(ei, Hh, AGGR);

    update_decode_kernel<<<nblk, 256, 0, stream>>>(Hh, AGGR, w_b1, w_b2, d_b1,
                                                   d_w2, d_b2, Ww1p, Ww2p, Dw1p, out);
}
